// LorentzSelfAttention_39109972198027
// MI455X (gfx1250) — compile-verified, matrix operand order checked by reading, not run on hardware
//
#include <hip/hip_runtime.h>
#include <hip/hip_bf16.h>
#include <math.h>

typedef __attribute__((ext_vector_type(16))) _Float16 v16h;
typedef __attribute__((ext_vector_type(8)))  float    v8f;
typedef __attribute__((ext_vector_type(4))) unsigned int u32x4;
typedef __attribute__((ext_vector_type(8))) int         i32x8;
typedef __attribute__((ext_vector_type(4))) int         i32x4;

#define TINYF 1e-15f
#define EPSF  1e-6f
#define T_ 512
#define D_ 512
#define H_ 8
#define N_ 64
#define A_ 65
#define QLD 68    // f32 Q row stride (padded)
#define HLD 96    // f16 head-vector stride, padded to 3*32 for guard-free WMMA K-loop
#define ULD 80
#define ZLD 544   // 17*32, covers H*A=520 with zero padding

#if __has_builtin(__builtin_amdgcn_tensor_load_to_lds) && \
    __has_builtin(__builtin_amdgcn_s_wait_tensorcnt)
#define HAVE_TDM 1
#else
#define HAVE_TDM 0
#endif

// ---------------------------------------------------------------------------
// WMMA fragment helpers (CDNA5 wave32 16x16x32 f16 layouts, ISA 7.12.2)
// A (16x32, 16-bit): lane L: row m = L&15, half = L>>4;
//   element e -> K = 8*half + (e&7) + ((e>>3)<<4)
// B (32x16, 16-bit): lane L: col n = L&15; same K pattern per element.
// D (16x16 f32): vgpr r -> M = r + 8*(L>>4), N = L&15.
// ---------------------------------------------------------------------------
__device__ __forceinline__ int fragK(int e, int half) {
  return 8 * half + (e & 7) + ((e >> 3) << 4);
}

__device__ __forceinline__ v16h fragA_f32(const float* p, int ld, int row, int k0, int half) {
  const float* r = p + (size_t)row * ld + k0 + 8 * half;
  v16h f;
#pragma unroll
  for (int e = 0; e < 8; ++e)  f[e] = (_Float16)r[e];
#pragma unroll
  for (int e = 8; e < 16; ++e) f[e] = (_Float16)r[e + 8];
  return f;
}

__device__ __forceinline__ v16h fragA_f16(const _Float16* p, int ld, int row, int k0, int half) {
  const _Float16* r = p + (size_t)row * ld + k0 + 8 * half;
  v16h f;
#pragma unroll
  for (int e = 0; e < 8; ++e)  f[e] = r[e];
#pragma unroll
  for (int e = 8; e < 16; ++e) f[e] = r[e + 8];
  return f;
}

__device__ __forceinline__ v16h fragB_f16(const _Float16* p, int ld, int col, int k0, int half) {
  v16h f;
#pragma unroll
  for (int e = 0; e < 16; ++e)
    f[e] = p[(size_t)(k0 + fragK(e, half)) * ld + col];
  return f;
}

// B-fragment from an LDS-staged column panel: lw[k][n], n = 0..15
__device__ __forceinline__ v16h fragB_lds(const float* lw, int col, int k0, int half) {
  v16h f;
#pragma unroll
  for (int e = 0; e < 16; ++e)
    f[e] = (_Float16)lw[(k0 + fragK(e, half)) * 16 + col];
  return f;
}

// ---------------------------------------------------------------------------
// Tensor Data Mover: stage a 2D tile (16 elems wide x tile_d1 rows) of a
// row-major f32 tensor (row length tensor_d0, row stride d0_stride, tensor_d1
// rows) into LDS. OOB rows (>= tensor_d1) read as zero.
// D# layout per CDNA5 ISA. 6-arg builtin form
// (clang-23/therock-10.0); falls back to a cooperative wave fill.
// ---------------------------------------------------------------------------
__device__ __forceinline__ void stage_panel_f32(float* lw, const float* gsrc,
                                                unsigned tensor_d0, unsigned tensor_d1,
                                                unsigned d0_stride,
                                                unsigned tile_d1, int lane) {
#if HAVE_TDM
  unsigned long long ga = (unsigned long long)(const void*)gsrc;
  unsigned lds_off = (unsigned)(unsigned long long)(const void*)lw;  // LDS aperture low bits
  u32x4 g0;
  g0[0] = 1u;                                             // count=1, user descriptor
  g0[1] = lds_off;                                        // lds_addr
  g0[2] = (unsigned)(ga & 0xffffffffull);                 // global_addr[31:0]
  g0[3] = (unsigned)((ga >> 32) & 0x1ffffffull) | (2u << 30);  // addr[56:32] | type=2
  i32x8 g1;
  g1[0] = (int)(2u << 16);                                // wg_mask=0, data_size=2 (4B)
  g1[1] = (int)((tensor_d0 & 0xffffu) << 16);             // tensor_dim0[15:0]
  g1[2] = (int)(((tensor_d0 >> 16) & 0xffffu) | ((tensor_d1 & 0xffffu) << 16));
  g1[3] = (int)(((tensor_d1 >> 16) & 0xffffu) | (16u << 16));  // tile_dim0 = 16
  g1[4] = (int)(tile_d1 & 0xffffu);                       // tile_dim1; tile_dim2=0
  g1[5] = (int)d0_stride;                                 // tensor_dim0_stride[31:0]
  g1[6] = 0;                                              // stride0[47:32], dim1_stride lo
  g1[7] = 0;
  i32x4 z4 = {0, 0, 0, 0};
  i32x8 z8 = {0, 0, 0, 0, 0, 0, 0, 0};
  __builtin_amdgcn_tensor_load_to_lds(g0, g1, z4, z4, z8, 0);
  __builtin_amdgcn_s_wait_tensorcnt(0);
#else
  for (unsigned idx = (unsigned)lane; idx < 16u * tile_d1; idx += 32u) {
    unsigned k = idx >> 4, n = idx & 15u;
    lw[idx] = (k < tensor_d1) ? gsrc[(size_t)k * d0_stride + n] : 0.0f;
  }
#endif
  __syncthreads();
}

// ---------------------------------------------------------------------------
// Kernel 1: QKV projection GEMMs. grid (32, 96), block 32 (one wave per tile).
// B panel W[:, j0:j0+16] staged to LDS by the TDM, then WMMA over K=512.
// ---------------------------------------------------------------------------
__global__ void k_gemm_qkv(const float* __restrict__ x,
                           const float* __restrict__ Wq, const float* __restrict__ Wk,
                           const float* __restrict__ Wv,
                           float* __restrict__ uq, float* __restrict__ uk, float* __restrict__ uv) {
  __shared__ float lw[D_ * 16];    // 32 KB column panel
  int lane = threadIdx.x & 31;
  int mrow = lane & 15, half = lane >> 4;
  int i0 = blockIdx.x * 16;
  int which = blockIdx.y >> 5;
  int j0 = (blockIdx.y & 31) * 16;
  const float* W = (which == 0) ? Wq : (which == 1) ? Wk : Wv;
  float* out = (which == 0) ? uq : (which == 1) ? uk : uv;

  stage_panel_f32(lw, W + j0, D_, D_, D_, D_, lane);

  v8f acc = {};
  for (int k0 = 0; k0 < D_; k0 += 32) {
    v16h a = fragA_f32(x, D_, i0 + mrow, k0, half);
    v16h b = fragB_lds(lw, mrow, k0, half);
    acc = __builtin_amdgcn_wmma_f32_16x16x32_f16(false, a, false, b, (short)0, acc, false, false);
  }
#pragma unroll
  for (int r = 0; r < 8; ++r)
    out[(size_t)(i0 + r + 8 * half) * D_ + j0 + mrow] = acc[r];
}

// ---------------------------------------------------------------------------
// Kernel 2: bias + RoPE(q,k) + lift onto hyperboloid + staging copies
// grid (32, 3), block 128; idx -> (h,t); blockIdx.y = 0:q 1:k 2:v
// ---------------------------------------------------------------------------
__global__ void k_lift(const float* __restrict__ uq, const float* __restrict__ uk,
                       const float* __restrict__ uv,
                       const float* __restrict__ bq, const float* __restrict__ bk,
                       const float* __restrict__ bv,
                       float* __restrict__ Qf, _Float16* __restrict__ Qng,
                       _Float16* __restrict__ Kh, _Float16* __restrict__ Vh,
                       float* __restrict__ mQQ, float* __restrict__ mVV) {
  int idx = blockIdx.x * blockDim.x + threadIdx.x;
  if (idx >= H_ * T_) return;
  int which = blockIdx.y;
  int h = idx / T_, t = idx % T_;
  const float* U  = (which == 0) ? uq : (which == 1) ? uk : uv;
  const float* bb = (which == 0) ? bq : (which == 1) ? bk : bv;

  float u[N_];
#pragma unroll 4
  for (int n = 0; n < N_; ++n) u[n] = U[(size_t)t * (H_ * N_) + h * N_ + n] + bb[h * N_ + n];

  if (which < 2) {  // RoPE
    float ur[N_];
#pragma unroll 4
    for (int i = 0; i < N_ / 2; ++i) {
      float f = powf(10000.0f, -(float)i / (float)(N_ / 2));
      float ang = (float)t * f;
      float c = cosf(ang), s = sinf(ang);
      ur[i]          = u[i] * c - u[i + N_ / 2] * s;
      ur[i + N_ / 2] = u[i + N_ / 2] * c + u[i] * s;
    }
    for (int n = 0; n < N_; ++n) u[n] = ur[n];
  }

  // lift = exp_map at origin: y = (cosh|u|, sinh|u|/|u| * u), then project
  float ss = 0.f;
  for (int n = 0; n < N_; ++n) ss += u[n] * u[n];
  float vn = sqrtf(fmaxf(ss, TINYF));
  float ch = coshf(vn), sh = sinhf(vn) / fmaxf(vn, TINYF);
  float y[A_];
  y[0] = ch;
  for (int n = 0; n < N_; ++n) y[n + 1] = sh * u[n];
  float xx = -y[0] * y[0];
  for (int e = 1; e < A_; ++e) xx += y[e] * y[e];
  float sc = sqrtf(fmaxf(fabsf(xx), TINYF));
  for (int e = 0; e < A_; ++e) y[e] /= sc;
  y[0] = fabsf(y[0]);
  float mk = -y[0] * y[0];
  for (int e = 1; e < A_; ++e) mk += y[e] * y[e];

  size_t rh = (size_t)h * T_ + t;
  if (which == 0) {
    for (int e = 0; e < A_; ++e)  Qf[rh * QLD + e] = y[e];
    for (int e = A_; e < QLD; ++e) Qf[rh * QLD + e] = 0.f;
    Qng[rh * HLD + 0] = (_Float16)y[0];               // sign-folded Minkowski form
    for (int e = 1; e < A_; ++e)  Qng[rh * HLD + e] = (_Float16)(-y[e]);
    for (int e = A_; e < HLD; ++e) Qng[rh * HLD + e] = (_Float16)0.f;
    mQQ[rh] = mk;
  } else if (which == 1) {
    for (int e = 0; e < A_; ++e)  Kh[rh * HLD + e] = (_Float16)y[e];
    for (int e = A_; e < HLD; ++e) Kh[rh * HLD + e] = (_Float16)0.f;
  } else {
    for (int e = 0; e < A_; ++e)  Vh[rh * HLD + e] = (_Float16)y[e];
    for (int e = A_; e < HLD; ++e) Vh[rh * HLD + e] = (_Float16)0.f;
    mVV[rh] = mk;
  }
}

// ---------------------------------------------------------------------------
// Kernel 3: QK scores. alpha = dot(Qneg,K); S = -arccosh(max(alpha,1+eps))^2
// grid (32,32,8), block 32
// ---------------------------------------------------------------------------
__global__ void k_scores(const _Float16* __restrict__ Qng, const _Float16* __restrict__ Kh,
                         float* __restrict__ S) {
  int lane = threadIdx.x & 31;
  int mrow = lane & 15, half = lane >> 4;
  int i0 = blockIdx.x * 16, j0 = blockIdx.y * 16, h = blockIdx.z;
  const _Float16* Qp = Qng + (size_t)h * T_ * HLD;
  const _Float16* Kp = Kh  + (size_t)h * T_ * HLD;
  v8f acc = {};
#pragma unroll
  for (int k0 = 0; k0 < HLD; k0 += 32) {
    v16h a = fragA_f16(Qp, HLD, i0 + mrow, k0, half);
    v16h b = fragA_f16(Kp, HLD, j0 + mrow, k0, half);  // B^T stored row-major
    acc = __builtin_amdgcn_wmma_f32_16x16x32_f16(false, a, false, b, (short)0, acc, false, false);
  }
  float* Sp = S + (size_t)h * T_ * T_;
#pragma unroll
  for (int r = 0; r < 8; ++r) {
    float d = acoshf(fmaxf(acc[r], 1.0f + EPSF));
    Sp[(size_t)(i0 + r + 8 * half) * T_ + j0 + mrow] = -d * d;  // TAU = 1
  }
}

// ---------------------------------------------------------------------------
// Kernel 4: row softmax in place. grid (512, 8), block 256
// ---------------------------------------------------------------------------
__global__ void k_softmax(float* __restrict__ S) {
  int i = blockIdx.x, h = blockIdx.y, tid = threadIdx.x;
  float* row = S + ((size_t)h * T_ + i) * T_;
  __shared__ float red[256];
  float a0 = row[tid], a1 = row[tid + 256];
  red[tid] = fmaxf(a0, a1);
  __syncthreads();
  for (int s = 128; s > 0; s >>= 1) { if (tid < s) red[tid] = fmaxf(red[tid], red[tid + s]); __syncthreads(); }
  float mx = red[0];
  __syncthreads();
  float e0 = expf(a0 - mx), e1 = expf(a1 - mx);
  red[tid] = e0 + e1;
  __syncthreads();
  for (int s = 128; s > 0; s >>= 1) { if (tid < s) red[tid] += red[tid + s]; __syncthreads(); }
  float inv = 1.0f / red[0];
  row[tid] = e0 * inv;
  row[tid + 256] = e1 * inv;
}

// ---------------------------------------------------------------------------
// Kernel 5: QV pairwise log-map scalars.
// alpha_v = dot(Qneg,V); c = arccosh(a)/sqrt(max(<u,u>,TINY));
// <u,u> = mVV[j] + a^2*mQQ[i] - 2a^2.  w = attn*c (f16); P = w*a (f32).
// grid (32,32,8), block 32
// ---------------------------------------------------------------------------
__global__ void k_pv(const _Float16* __restrict__ Qng, const _Float16* __restrict__ Vh,
                     const float* __restrict__ attn, const float* __restrict__ mQQ,
                     const float* __restrict__ mVV,
                     _Float16* __restrict__ wh, float* __restrict__ P) {
  int lane = threadIdx.x & 31;
  int mrow = lane & 15, half = lane >> 4;
  int i0 = blockIdx.x * 16, j0 = blockIdx.y * 16, h = blockIdx.z;
  const _Float16* Qp = Qng + (size_t)h * T_ * HLD;
  const _Float16* Vp = Vh  + (size_t)h * T_ * HLD;
  v8f acc = {};
#pragma unroll
  for (int k0 = 0; k0 < HLD; k0 += 32) {
    v16h a = fragA_f16(Qp, HLD, i0 + mrow, k0, half);
    v16h b = fragA_f16(Vp, HLD, j0 + mrow, k0, half);
    acc = __builtin_amdgcn_wmma_f32_16x16x32_f16(false, a, false, b, (short)0, acc, false, false);
  }
  int j = j0 + mrow;
  float mvv = mVV[(size_t)h * T_ + j];
#pragma unroll
  for (int r = 0; r < 8; ++r) {
    int i = i0 + r + 8 * half;
    float av = acc[r];
    float dist = acoshf(fmaxf(av, 1.0f + EPSF));
    float muu = mvv + av * av * mQQ[(size_t)h * T_ + i] - 2.0f * av * av;
    float un = sqrtf(fmaxf(muu, TINYF));
    float c = dist / fmaxf(un, TINYF);
    size_t o = ((size_t)h * T_ + i) * T_ + j;
    float w = attn[o] * c;
    wh[o] = (_Float16)w;
    P[o] = w * av;
  }
}

// ---------------------------------------------------------------------------
// Kernel 6: g[h][i] = sum_j P[h][i][j]. grid (512, 8), block 128
// ---------------------------------------------------------------------------
__global__ void k_gred(const float* __restrict__ P, float* __restrict__ g) {
  int i = blockIdx.x, h = blockIdx.y, tid = threadIdx.x;
  const float* row = P + ((size_t)h * T_ + i) * T_;
  __shared__ float red[128];
  red[tid] = row[tid] + row[tid + 128] + row[tid + 256] + row[tid + 384];
  __syncthreads();
  for (int s = 64; s > 0; s >>= 1) { if (tid < s) red[tid] += red[tid + s]; __syncthreads(); }
  if (tid == 0) g[(size_t)h * T_ + i] = red[0];
}

// ---------------------------------------------------------------------------
// Kernel 7: U = W @ V per head (512 x 512 x 65->80). grid (32, 5, 8), block 32
// ---------------------------------------------------------------------------
__global__ void k_wv(const _Float16* __restrict__ wh, const _Float16* __restrict__ Vh,
                     float* __restrict__ Uo) {
  int lane = threadIdx.x & 31;
  int mrow = lane & 15, half = lane >> 4;
  int i0 = blockIdx.x * 16, e0 = blockIdx.y * 16, h = blockIdx.z;
  const _Float16* Ap = wh + (size_t)h * T_ * T_;
  const _Float16* Bp = Vh + (size_t)h * T_ * HLD;
  v8f acc = {};
  for (int k0 = 0; k0 < T_; k0 += 32) {
    v16h a = fragA_f16(Ap, T_, i0 + mrow, k0, half);
    v16h b = fragB_f16(Bp, HLD, e0 + mrow, k0, half);
    acc = __builtin_amdgcn_wmma_f32_16x16x32_f16(false, a, false, b, (short)0, acc, false, false);
  }
  float* Up = Uo + (size_t)h * T_ * ULD;
#pragma unroll
  for (int r = 0; r < 8; ++r)
    Up[(size_t)(i0 + r + 8 * half) * ULD + e0 + mrow] = acc[r];
}

// ---------------------------------------------------------------------------
// Kernel 8: u = U + g*Q ; exp_map(Q,u) ; project ; log_map(anchor,.) -> Z (f16)
// grid (64), block 64; idx -> (h,t)
// ---------------------------------------------------------------------------
__global__ void k_ymap(const float* __restrict__ Uo, const float* __restrict__ g,
                       const float* __restrict__ Qf, const float* __restrict__ anchor,
                       _Float16* __restrict__ Zh) {
  int idx = blockIdx.x * blockDim.x + threadIdx.x;
  if (idx >= H_ * T_) return;
  int h = idx / T_, t = idx % T_;
  size_t rh = (size_t)h * T_ + t;

  float u[A_], q[A_];
  float gg = g[rh];
  for (int e = 0; e < A_; ++e) {
    q[e] = Qf[rh * QLD + e];
    u[e] = Uo[rh * ULD + e] + gg * q[e];
  }
  float mm = -u[0] * u[0];
  for (int e = 1; e < A_; ++e) mm += u[e] * u[e];
  float vn = sqrtf(fmaxf(mm, TINYF));
  float ch = coshf(vn), sh = sinhf(vn) / fmaxf(vn, TINYF);
  float y[A_];
  for (int e = 0; e < A_; ++e) y[e] = ch * q[e] + sh * u[e];
  float xx = -y[0] * y[0];
  for (int e = 1; e < A_; ++e) xx += y[e] * y[e];
  float sc = sqrtf(fmaxf(fabsf(xx), TINYF));
  for (int e = 0; e < A_; ++e) y[e] /= sc;
  y[0] = fabsf(y[0]);

  float ap[A_];
  for (int e = 0; e < A_; ++e) ap[e] = anchor[e];
  float al = ap[0] * y[0];
  for (int e = 1; e < A_; ++e) al -= ap[e] * y[e];   // alpha = -mink(anchor, y)
  float dist = acoshf(fmaxf(al, 1.0f + EPSF));
  float uu[A_];
  for (int e = 0; e < A_; ++e) uu[e] = y[e] + al * ap[e];
  float m2 = -uu[0] * uu[0];
  for (int e = 1; e < A_; ++e) m2 += uu[e] * uu[e];
  float un = sqrtf(fmaxf(m2, TINYF));
  float cc = dist / fmaxf(un, TINYF);

  for (int e = 0; e < A_; ++e)
    Zh[(size_t)t * ZLD + h * A_ + e] = (_Float16)(cc * uu[e]);
  if (h == 0)  // zero the K-padding columns once per row
    for (int e = H_ * A_; e < ZLD; ++e) Zh[(size_t)t * ZLD + e] = (_Float16)0.f;
}

// ---------------------------------------------------------------------------
// Kernel 9: out = Z @ Wo + bo. grid (32, 32), block 32. K = 544; the TDM's
// OOB-row-reads-zero handles rows 520..543 (tensor_dim1 = 520, tile_dim1 = 544).
// ---------------------------------------------------------------------------
__global__ void k_out(const _Float16* __restrict__ Zh, const float* __restrict__ Wo,
                      const float* __restrict__ bo, float* __restrict__ out) {
  __shared__ float lw[ZLD * 16];   // 34 KB column panel (zero-filled tail)
  int lane = threadIdx.x & 31;
  int mrow = lane & 15, half = lane >> 4;
  int i0 = blockIdx.x * 16, j0 = blockIdx.y * 16;

  stage_panel_f32(lw, Wo + j0, D_, H_ * A_, D_, ZLD, lane);

  v8f acc = {};
  for (int k0 = 0; k0 < ZLD; k0 += 32) {
    v16h a = fragA_f16(Zh, ZLD, i0 + mrow, k0, half);
    v16h b = fragB_lds(lw, mrow, k0, half);
    acc = __builtin_amdgcn_wmma_f32_16x16x32_f16(false, a, false, b, (short)0, acc, false, false);
  }
  float bias = bo[j0 + mrow];
#pragma unroll
  for (int r = 0; r < 8; ++r)
    out[(size_t)(i0 + r + 8 * half) * D_ + j0 + mrow] = acc[r] + bias;
}

// ---------------------------------------------------------------------------
extern "C" void kernel_launch(void* const* d_in, const int* in_sizes, int n_in,
                              void* d_out, int out_size, void* d_ws, size_t ws_size,
                              hipStream_t stream) {
  (void)in_sizes; (void)n_in; (void)out_size; (void)ws_size;
  const float* x   = (const float*)d_in[0];
  const float* Wq  = (const float*)d_in[1];
  const float* bq  = (const float*)d_in[2];
  const float* Wk  = (const float*)d_in[3];
  const float* bk  = (const float*)d_in[4];
  const float* Wv  = (const float*)d_in[5];
  const float* bv  = (const float*)d_in[6];
  const float* Wo  = (const float*)d_in[7];
  const float* bo  = (const float*)d_in[8];
  const float* anc = (const float*)d_in[9];
  float* out = (float*)d_out;

  char* ws = (char*)d_ws;
  size_t off = 0;
  auto take = [&](size_t bytes) -> char* {
    char* p = ws + off;
    off = (off + bytes + 255) & ~(size_t)255;
    return p;
  };
  float*    uq   = (float*)take((size_t)T_ * D_ * 4);
  float*    uk   = (float*)take((size_t)T_ * D_ * 4);
  float*    uv   = (float*)take((size_t)T_ * D_ * 4);
  float*    Qf   = (float*)take((size_t)H_ * T_ * QLD * 4);
  _Float16* Qng  = (_Float16*)take((size_t)H_ * T_ * HLD * 2);
  _Float16* Kh   = (_Float16*)take((size_t)H_ * T_ * HLD * 2);
  _Float16* Vh   = (_Float16*)take((size_t)H_ * T_ * HLD * 2);
  float*    mQQ  = (float*)take((size_t)H_ * T_ * 4);
  float*    mVV  = (float*)take((size_t)H_ * T_ * 4);
  float*    S    = (float*)take((size_t)H_ * T_ * T_ * 4);   // scores -> attn in place
  _Float16* wh   = (_Float16*)take((size_t)H_ * T_ * T_ * 2);
  float*    P    = (float*)take((size_t)H_ * T_ * T_ * 4);
  float*    g    = (float*)take((size_t)H_ * T_ * 4);
  float*    Uo   = (float*)take((size_t)H_ * T_ * ULD * 4);
  _Float16* Zh   = (_Float16*)take((size_t)T_ * ZLD * 2);

  k_gemm_qkv<<<dim3(T_ / 16, 3 * D_ / 16), 32, 0, stream>>>(x, Wq, Wk, Wv, uq, uk, uv);
  k_lift<<<dim3((H_ * T_) / 128, 3), 128, 0, stream>>>(uq, uk, uv, bq, bk, bv,
                                                       Qf, Qng, Kh, Vh, mQQ, mVV);
  k_scores<<<dim3(T_ / 16, T_ / 16, H_), 32, 0, stream>>>(Qng, Kh, S);
  k_softmax<<<dim3(T_, H_), 256, 0, stream>>>(S);
  k_pv<<<dim3(T_ / 16, T_ / 16, H_), 32, 0, stream>>>(Qng, Vh, S, mQQ, mVV, wh, P);
  k_gred<<<dim3(T_, H_), 128, 0, stream>>>(P, g);
  k_wv<<<dim3(T_ / 16, ULD / 16, H_), 32, 0, stream>>>(wh, Vh, Uo);
  k_ymap<<<dim3((H_ * T_) / 64), 64, 0, stream>>>(Uo, g, Qf, anc, Zh);
  k_out<<<dim3(T_ / 16, D_ / 16), 32, 0, stream>>>(Zh, Wo, bo, out);
}
